// EQUpdate_24833500905740
// MI455X (gfx1250) — matrix operand order repaired and checked by reading, not compiled, not run on hardware
//
#include <hip/hip_runtime.h>
#include <hip/hip_bf16.h>
#include <cstdint>

// Problem constants (from reference)
#define N_NODES 10000
#define N_EDGES 640000
#define HDIM    128
// COORD_RANGE / 100 = (12/8)/100
#define OUT_SCALE 0.015f

typedef _Float16 v16h __attribute__((ext_vector_type(16)));
typedef _Float16 v8h  __attribute__((ext_vector_type(8)));
typedef float    v8f  __attribute__((ext_vector_type(8)));

union F16x16 { v16h v; v8h h[2]; _Float16 e[16]; };

// ---------------------------------------------------------------------------
// Kernel 1: pack W1/W2 (f32 [K,128]) into f16 WMMA B-fragments.
// B-fragment (32x16, f16) per-lane layout: lane<16 -> col n=lane, halves 0-7 = K 0-7, 8-15 = K 16-23;
// lane>=16 -> col n=lane-16, halves 0-7 = K 8-15, 8-15 = K 24-31. Fragment storage is
// frag*512 + lane*16 + j halves, so the GEMM kernel loads one v16h per lane.
// Also stash b1, W1 rows 256/257, b2, W3 as f32 vectors.
// ---------------------------------------------------------------------------
__global__ void pack_weights_kernel(const float* __restrict__ W1,
                                    const float* __restrict__ b1,
                                    const float* __restrict__ W2,
                                    const float* __restrict__ b2,
                                    const float* __restrict__ W3,
                                    _Float16* __restrict__ pW1,
                                    _Float16* __restrict__ pW2,
                                    float* __restrict__ cvec) {
  int tid = blockIdx.x * blockDim.x + threadIdx.x;
  if (tid < 32768) {                       // W1: 8 nt x 8 kt fragments
    int frag = tid >> 9, lane = (tid >> 4) & 31, j = tid & 15;
    int nt = frag >> 3, kt = frag & 7;
    int k = kt * 32 + ((lane < 16) ? 0 : 8) + (j & 7) + ((j & 8) << 1);
    int n = nt * 16 + (lane & 15);
    pW1[tid] = (_Float16)W1[k * HDIM + n];
  } else if (tid < 49152) {                // W2: 8 nt x 4 kt fragments
    int t = tid - 32768;
    int frag = t >> 9, lane = (t >> 4) & 31, j = t & 15;
    int nt = frag >> 2, kt = frag & 3;
    int k = kt * 32 + ((lane < 16) ? 0 : 8) + (j & 7) + ((j & 8) << 1);
    int n = nt * 16 + (lane & 15);
    pW2[t] = (_Float16)W2[k * HDIM + n];
  } else if (tid < 49152 + 640) {          // f32 vectors: b1, W1r256, W1r257, b2, W3
    int t = tid - 49152, w = t >> 7, n = t & 127;
    float v;
    if      (w == 0) v = b1[n];
    else if (w == 1) v = W1[256 * HDIM + n];
    else if (w == 2) v = W1[257 * HDIM + n];
    else if (w == 3) v = b2[n];
    else             v = W3[n];
    cvec[t] = v;
  }
}

// ---------------------------------------------------------------------------
// Kernel 2: out = x
// ---------------------------------------------------------------------------
__global__ void init_out_kernel(const float* __restrict__ x,
                                float* __restrict__ out, int n) {
  int i = blockIdx.x * blockDim.x + threadIdx.x;
  if (i < n) out[i] = x[i];
}

// ---------------------------------------------------------------------------
// Build an f16 A-fragment slice from 8+8 consecutive f32 features:
// halves 0..7  = p[0..7]   (K base+0..7  for this lane group)
// halves 8..15 = p[16..23] (K base+16..23)
// ---------------------------------------------------------------------------
__device__ __forceinline__ v16h load_cvt_a(const float* __restrict__ p) {
  float4 f0 = *(const float4*)(p);
  float4 f1 = *(const float4*)(p + 4);
  float4 f2 = *(const float4*)(p + 16);
  float4 f3 = *(const float4*)(p + 20);
  F16x16 r;
  r.e[0]  = (_Float16)f0.x; r.e[1]  = (_Float16)f0.y;
  r.e[2]  = (_Float16)f0.z; r.e[3]  = (_Float16)f0.w;
  r.e[4]  = (_Float16)f1.x; r.e[5]  = (_Float16)f1.y;
  r.e[6]  = (_Float16)f1.z; r.e[7]  = (_Float16)f1.w;
  r.e[8]  = (_Float16)f2.x; r.e[9]  = (_Float16)f2.y;
  r.e[10] = (_Float16)f2.z; r.e[11] = (_Float16)f2.w;
  r.e[12] = (_Float16)f3.x; r.e[13] = (_Float16)f3.y;
  r.e[14] = (_Float16)f3.z; r.e[15] = (_Float16)f3.w;
  return r.v;
}

// SiLU via fast hardware reciprocal (v_rcp_f32) instead of IEEE division:
// avoids the v_div_scale/v_div_fmas/v_div_fixup expansion seen in round 1.
__device__ __forceinline__ float silu(float f) {
  return f * __builtin_amdgcn_rcpf(1.0f + __expf(-f));
}

// tanh(x) = 1 - 2/(exp(2x)+1); saturates correctly at +/-inf of the exp.
__device__ __forceinline__ float fast_tanh(float x) {
  return 1.0f - 2.0f * __builtin_amdgcn_rcpf(__expf(2.0f * x) + 1.0f);
}

// ---------------------------------------------------------------------------
// Kernel 3: main edge MLP. One wave per 16-edge tile; 8 waves per block.
// Layer1: C[16x128] = [h_row|h_col][16x256] x W1k + (b1 + d*w256 + dorg*w257)
// Layer2: C[16x128] = silu(L1)[16x128] x W2 + b2
// Layer3: per-lane dot with W3 + shfl_xor reduce; tanh; atomic scatter-add.
// ---------------------------------------------------------------------------
__global__ __launch_bounds__(256) void egnn_edge_kernel(
    const float* __restrict__ h, const long long* __restrict__ edges,
    const float* __restrict__ coord_diff, const float* __restrict__ dist,
    const float* __restrict__ dorg,
    const _Float16* __restrict__ pW1, const _Float16* __restrict__ pW2,
    const float* __restrict__ cvec, float* __restrict__ out) {
  // per-wave activation staging: 16 rows x 128 f16, stride 136 halves
  // (272 B = 68 banks -> rotates 4 banks/row, avoids column-store conflicts)
  __shared__ __align__(32) _Float16 act[8][16 * 136];

  const int lane = threadIdx.x & 31;
  const int wave = threadIdx.x >> 5;
  const int tile = blockIdx.x * 8 + wave;
  const int e0   = tile * 16;
  const int m    = lane & 15;   // edge-in-tile this lane's A-row covers
  const int hi   = lane >> 4;   // lane group: K-lo (0) / K-hi (1)

  const int ebase = e0 + m;
  const int rnode = (int)edges[ebase];            // edges[0][e]
  const int cnode = (int)edges[N_EDGES + ebase];  // edges[1][e]

  // ---- Layer-1 A fragments: K = [h[row](128) | h[col](128)] -------------
  F16x16 A1[8];
#pragma unroll
  for (int kt = 0; kt < 8; ++kt) {
    const int  node = (kt < 4) ? rnode : cnode;
    const int  base = (kt & 3) * 32 + hi * 8;
    A1[kt].v = load_cvt_a(h + node * HDIM + base);
  }

  // per-row scalar features for the C-init (rows m = v + hi*8)
  float dl[8], dg[8];
  {
    const float4* dp = (const float4*)(dist + e0 + hi * 8);
    float4 a0 = dp[0], a1 = dp[1];
    dl[0] = a0.x; dl[1] = a0.y; dl[2] = a0.z; dl[3] = a0.w;
    dl[4] = a1.x; dl[5] = a1.y; dl[6] = a1.z; dl[7] = a1.w;
    const float4* gp = (const float4*)(dorg + e0 + hi * 8);
    float4 b0 = gp[0], b1 = gp[1];
    dg[0] = b0.x; dg[1] = b0.y; dg[2] = b0.z; dg[3] = b0.w;
    dg[4] = b1.x; dg[5] = b1.y; dg[6] = b1.z; dg[7] = b1.w;
  }

  const float* b1c  = cvec;
  const float* r256 = cvec + 128;
  const float* r257 = cvec + 256;
  const float* b2c  = cvec + 384;
  const float* w3c  = cvec + 512;
  const int n = m;  // C-matrix column for this lane

  // ---- Layer 1: 8 N-tiles x 8 K-steps = 64 WMMA --------------------------
  for (int nt = 0; nt < 8; ++nt) {
    const float bb = b1c[nt * 16 + n];
    const float w6 = r256[nt * 16 + n];
    const float w7 = r257[nt * 16 + n];
    v8f c;
#pragma unroll
    for (int v = 0; v < 8; ++v) c[v] = fmaf(dl[v], w6, fmaf(dg[v], w7, bb));
#pragma unroll
    for (int kt = 0; kt < 8; ++kt) {
      v16h b = *(const v16h*)(pW1 + ((nt * 8 + kt) << 9) + (lane << 4));
      c = __builtin_amdgcn_wmma_f32_16x16x32_f16(false, A1[kt].v, false, b,
                                                 (short)0, c, false, false);
    }
#pragma unroll
    for (int v = 0; v < 8; ++v) {
      float f = silu(c[v]);
      act[wave][(v + hi * 8) * 136 + nt * 16 + n] = (_Float16)f;
    }
  }

  __syncthreads();  // LDS stores -> cross-lane A-fragment loads

  // ---- Layer-2 A fragments from LDS (row-major act, A-matrix layout) -----
  F16x16 A2[4];
#pragma unroll
  for (int kt = 0; kt < 4; ++kt) {
    const int off = kt * 32 + hi * 8;
    A2[kt].h[0] = *(const v8h*)(&act[wave][m * 136 + off]);
    A2[kt].h[1] = *(const v8h*)(&act[wave][m * 136 + off + 16]);
  }

  // ---- Layer 2 (32 WMMA) + layer-3 partial dot ---------------------------
  float part[8];
#pragma unroll
  for (int v = 0; v < 8; ++v) part[v] = 0.0f;

  for (int nt = 0; nt < 8; ++nt) {
    const float bb  = b2c[nt * 16 + n];
    const float w3n = w3c[nt * 16 + n];
    v8f c;
#pragma unroll
    for (int v = 0; v < 8; ++v) c[v] = bb;
#pragma unroll
    for (int kt = 0; kt < 4; ++kt) {
      v16h b = *(const v16h*)(pW2 + ((nt * 4 + kt) << 9) + (lane << 4));
      c = __builtin_amdgcn_wmma_f32_16x16x32_f16(false, A2[kt].v, false, b,
                                                 (short)0, c, false, false);
    }
#pragma unroll
    for (int v = 0; v < 8; ++v) {
      part[v] = fmaf(silu(c[v]), w3n, part[v]);
    }
  }

  // ---- reduce over the 16 N-lanes (C layout: lane = n, vgpr = m) ---------
#pragma unroll
  for (int mask = 1; mask < 16; mask <<= 1) {
#pragma unroll
    for (int v = 0; v < 8; ++v) part[v] += __shfl_xor(part[v], mask, 32);
  }
  // lanes 0-15 hold s[0..7], lanes 16-31 hold s[8..15]

  const int sub = lane & 15;
  if (sub < 8) {
    const int mm = hi * 8 + sub;
    const int e  = e0 + mm;
    const float t = fast_tanh(part[sub]) * OUT_SCALE;
    const int r = (int)edges[e];
    const float c0 = coord_diff[e * 3 + 0];
    const float c1 = coord_diff[e * 3 + 1];
    const float c2 = coord_diff[e * 3 + 2];
    atomicAdd(&out[r * 3 + 0], c0 * t);
    atomicAdd(&out[r * 3 + 1], c1 * t);
    atomicAdd(&out[r * 3 + 2], c2 * t);
  }
}

// ---------------------------------------------------------------------------
extern "C" void kernel_launch(void* const* d_in, const int* in_sizes, int n_in,
                              void* d_out, int out_size, void* d_ws, size_t ws_size,
                              hipStream_t stream) {
  const float*     h          = (const float*)d_in[0];
  const float*     x          = (const float*)d_in[1];
  const long long* edges      = (const long long*)d_in[2];  // int64 per reference
  const float*     coord_diff = (const float*)d_in[3];
  const float*     dist       = (const float*)d_in[4];
  const float*     dorg       = (const float*)d_in[5];
  const float*     W1         = (const float*)d_in[6];
  const float*     b1         = (const float*)d_in[7];
  const float*     W2         = (const float*)d_in[8];
  const float*     b2         = (const float*)d_in[9];
  const float*     W3         = (const float*)d_in[10];
  float* out = (float*)d_out;

  char* ws = (char*)d_ws;
  _Float16* pW1 = (_Float16*)ws;                        // 65536 B
  _Float16* pW2 = (_Float16*)(ws + 65536);              // 32768 B
  float*    cvec = (float*)(ws + 65536 + 32768);        // 5*128*4 = 2560 B

  pack_weights_kernel<<<(49152 + 640 + 255) / 256, 256, 0, stream>>>(
      W1, b1, W2, b2, W3, pW1, pW2, cvec);
  init_out_kernel<<<(N_NODES * 3 + 255) / 256, 256, 0, stream>>>(
      x, out, N_NODES * 3);
  // 640000 edges / 16 per wave / 8 waves per block = 5000 blocks (exact)
  egnn_edge_kernel<<<N_EDGES / 16 / 8, 256, 0, stream>>>(
      h, edges, coord_diff, dist, dorg, pW1, pW2, cvec, out);
}
